// GCN_90031104459077
// MI455X (gfx1250) — matrix operand order repaired and checked by reading, not compiled, not run on hardware
//
#include <hip/hip_runtime.h>

typedef __attribute__((ext_vector_type(16))) _Float16 v16h;
typedef __attribute__((ext_vector_type(4)))  _Float16 v4h;
typedef __attribute__((ext_vector_type(2)))  _Float16 v2h;
typedef __attribute__((ext_vector_type(8)))  float    v8f;
typedef __attribute__((ext_vector_type(4)))  float    v4f;

// ---------------- weight f32 -> f16 conversion (one shot, tiny) ----------------
__global__ void cvt_f32_to_f16(const float* __restrict__ a, const float* __restrict__ b,
                               const float* __restrict__ c, int na, int nb, int nc,
                               _Float16* __restrict__ out) {
  int i = blockIdx.x * blockDim.x + threadIdx.x;
  int tot = na + nb + nc;
  if (i >= tot) return;
  float v;
  if (i < na)           v = a[i];
  else if (i < na + nb) v = b[i - na];
  else                  v = c[i - na - nb];
  out[i] = (_Float16)v;
}

// ---------------- zero fill ----------------
__global__ void zerof(float* __restrict__ p, size_t n) {
  size_t i = (size_t)blockIdx.x * blockDim.x + threadIdx.x;
  size_t stride = (size_t)gridDim.x * blockDim.x;
  for (; i < n; i += stride) p[i] = 0.0f;
}

// ---------------- GEMM: out_f16[M,NT*16] = act(A_f32[M,K]) @ W_f16[NT*16,K]^T ----------------
// W staged into LDS once per block (ds_load_b128 B-fragments, double-buffered across the
// column-tile loop so each s_wait_dscnt overlaps a WMMA in flight). Output stored as f16:
// it is only ever consumed by the SpMM gather, halving the dominant read stream.
template<int K, int NT, bool RELU>
__global__ void __launch_bounds__(128) gemm_wmma(const float* __restrict__ A,
                                                 const _Float16* __restrict__ W,
                                                 _Float16* __restrict__ out, int mtiles) {
  constexpr int NC = NT * 16;
  __shared__ _Float16 Wlds[NC * K];          // 64KB / 32KB / 16KB of the 320KB WGP LDS

  // --- cooperative stage: all 128 threads copy W -> LDS in 16B chunks ---
  {
    constexpr int CHUNKS = NC * K / 8;       // 8 halfs per 16B chunk
    const v4f* src = (const v4f*)W;
    v4f* dst = (v4f*)Wlds;
#pragma unroll 4
    for (int i = threadIdx.x; i < CHUNKS; i += 128) dst[i] = src[i];
  }
  __syncthreads();

  const int lane = threadIdx.x & 31;
  const int wave = threadIdx.x >> 5;
  const int mtile = blockIdx.x * 4 + wave;
  if (mtile >= mtiles) return;               // wave-uniform: EXEC all-ones for WMMA
  const int half = lane >> 4;                // lane group (0: lanes 0-15, 1: lanes 16-31)
  const int mr   = lane & 15;

  v8f acc[NT] = {};
  const float* arow = A + (size_t)(mtile * 16 + mr) * K;
  const _Float16* wb = Wlds + (size_t)mr * K + 16 * half;

#pragma unroll
  for (int kt = 0; kt < K / 32; ++kt) {
    const int kbase = kt * 32;
    // --- A fragment: 16x32 f16. Lane(half,mr): e[0..7] <- K=kbase+16*half+(0..7),
    //                                           e[8..15] <- K=kbase+16*half+8+(0..7)
    v16h afrag;
    {
      const float* p0 = arow + kbase + 16 * half;
      const float* p1 = p0 + 8;
      v4f x0 = *(const v4f*)p0;
      v4f x1 = *(const v4f*)(p0 + 4);
      v4f x2 = *(const v4f*)p1;
      v4f x3 = *(const v4f*)(p1 + 4);
#pragma unroll
      for (int j = 0; j < 4; ++j) {
        float e0 = x0[j], e1 = x1[j], e2 = x2[j], e3 = x3[j];
        if (RELU) {
          e0 = fmaxf(e0, 0.0f); e1 = fmaxf(e1, 0.0f);
          e2 = fmaxf(e2, 0.0f); e3 = fmaxf(e3, 0.0f);
        }
        afrag[j]      = (_Float16)e0;
        afrag[4 + j]  = (_Float16)e1;
        afrag[8 + j]  = (_Float16)e2;
        afrag[12 + j] = (_Float16)e3;
      }
    }
    // --- B fragments from LDS, double-buffered: lane(half,n=mr), e -> K=kbase+16*half+e
    const _Float16* wk = wb + kbase;
    v16h bcur = *(const v16h*)wk;
#pragma unroll
    for (int nt = 0; nt < NT; ++nt) {
      v16h bnext = (nt + 1 < NT) ? *(const v16h*)(wk + (size_t)(nt + 1) * 16 * K) : bcur;
      acc[nt] = __builtin_amdgcn_wmma_f32_16x16x32_f16(
          /*neg_a=*/false, afrag, /*neg_b=*/false, bcur,
          /*c_mod=*/(short)0, acc[nt], /*reuse_a=*/false, /*reuse_b=*/false);
      bcur = bnext;
    }
  }

  // --- D layout: VGPR v -> row 8*half+v, lane%16 -> col; store as f16
#pragma unroll
  for (int nt = 0; nt < NT; ++nt) {
    _Float16* op = out + (size_t)(mtile * 16) * NC + nt * 16 + mr;
#pragma unroll
    for (int v = 0; v < 8; ++v) {
      op[(size_t)(8 * half + v) * NC] = (_Float16)acc[nt][v];
    }
  }
}

// ---------------- SpMM, segmented: out_f32[row[e],:] += val[e] * Z_f16[col[e],:] ----------------
// edge_row is sorted -> one wave owns SEG consecutive edges, register-accumulates its
// FEAT/32-element slice in f32 while the row is unchanged (row test is wave-uniform),
// flushing f32 atomics only at row/segment boundaries. Gather is f16 (half the bytes).
template<int FEAT, int SEG>
__global__ void spmm_seg(const int* __restrict__ rows, const int* __restrict__ cols,
                         const float* __restrict__ vals, const _Float16* __restrict__ Z,
                         float* __restrict__ out, int E) {
  constexpr int VL = FEAT / 32;              // elems per lane (4 for H=128, 2 for F=64)
  const int lane = threadIdx.x & 31;
  const int wid  = (int)(((long long)blockIdx.x * blockDim.x + threadIdx.x) >> 5);
  long long e0 = (long long)wid * SEG;
  if (e0 >= E) return;
  const int eEnd = (int)((e0 + SEG < E) ? (e0 + SEG) : E);

  float acc[VL];
#pragma unroll
  for (int v = 0; v < VL; ++v) acc[v] = 0.0f;

  int curRow = rows[e0];
  for (int e = (int)e0; e < eEnd; ++e) {
    int r = rows[e];                          // same addr across lanes -> uniform
    if (r != curRow) {                        // wave-uniform branch
      float* o = out + (size_t)curRow * FEAT + lane * VL;
#pragma unroll
      for (int v = 0; v < VL; ++v) { atomicAdd(o + v, acc[v]); acc[v] = 0.0f; }
      curRow = r;
    }
    float w = vals[e];
    const _Float16* zp = Z + (size_t)cols[e] * FEAT + lane * VL;
    if constexpr (VL == 4) {
      v4h z = *(const v4h*)zp;
#pragma unroll
      for (int v = 0; v < 4; ++v) acc[v] += w * (float)z[v];
    } else {
      v2h z = *(const v2h*)zp;
#pragma unroll
      for (int v = 0; v < 2; ++v) acc[v] += w * (float)z[v];
    }
  }
  float* o = out + (size_t)curRow * FEAT + lane * VL;
#pragma unroll
  for (int v = 0; v < VL; ++v) atomicAdd(o + v, acc[v]);
}

// ---------------- row softmax over F=64, one wave32 per row ----------------
__global__ void softmax_rows64(float* __restrict__ out, int nrows) {
  int wid  = (int)(((long long)blockIdx.x * blockDim.x + threadIdx.x) >> 5);
  int lane = threadIdx.x & 31;
  if (wid >= nrows) return;
  float* row = out + (size_t)wid * 64;
  float a = row[lane];
  float b = row[lane + 32];
  float m = fmaxf(a, b);
#pragma unroll
  for (int s = 16; s > 0; s >>= 1) m = fmaxf(m, __shfl_xor(m, s, 32));
  float ea = __expf(a - m);
  float eb = __expf(b - m);
  float sum = ea + eb;
#pragma unroll
  for (int s = 16; s > 0; s >>= 1) sum += __shfl_xor(sum, s, 32);
  float inv = 1.0f / sum;
  row[lane]      = ea * inv;
  row[lane + 32] = eb * inv;
}

extern "C" void kernel_launch(void* const* d_in, const int* in_sizes, int n_in,
                              void* d_out, int out_size, void* d_ws, size_t ws_size,
                              hipStream_t stream) {
  (void)n_in; (void)out_size; (void)ws_size;
  const float* X    = (const float*)d_in[0];
  const int*   erow = (const int*)d_in[1];
  const int*   ecol = (const int*)d_in[2];
  const float* ev   = (const float*)d_in[3];
  const float* W0   = (const float*)d_in[4];
  const float* Wh0  = (const float*)d_in[5];
  const float* W1   = (const float*)d_in[6];

  const int C = 256, H = 128, F = 64;
  const int N = in_sizes[0] / C;     // 100000
  const int E = in_sizes[1];         // 1600000
  const int nW0 = in_sizes[4];       // H*C
  const int nWh = in_sizes[5];       // H*H
  const int nW1 = in_sizes[6];       // F*H

  // workspace layout:
  //   [f16 weights, 128KB pad] [bufH: N*H f16 activations] [bufF: N*H f32 spmm accumulator]
  _Float16* W0h  = (_Float16*)d_ws;
  _Float16* Wh0h = W0h + nW0;
  _Float16* W1h  = Wh0h + nWh;
  _Float16* bufH = (_Float16*)((char*)d_ws + 131072);
  float*    bufF = (float*)((char*)d_ws + 131072 + (size_t)N * H * sizeof(_Float16));
  float*    outp = (float*)d_out;

  // 0) weights -> f16
  {
    int tot = nW0 + nWh + nW1;
    cvt_f32_to_f16<<<(tot + 255) / 256, 256, 0, stream>>>(W0, Wh0, W1, nW0, nWh, nW1, W0h);
  }

  const int mtiles  = N / 16;               // 6250 (N divisible by 16)
  const int gblocks = (mtiles + 3) / 4;     // 4 waves (16-row stripes) per 128-thread block

  constexpr int SEG = 16;                   // edges per wave in segmented SpMM
  const int nwaves  = (E + SEG - 1) / SEG;
  const int sblocks = (nwaves + 7) / 8;     // 8 waves per 256-thread block

  // 1) bufH = f16( X @ W0^T )
  gemm_wmma<256, 8, false><<<gblocks, 128, 0, stream>>>(X, W0h, bufH, mtiles);
  // 2) bufF = A_hat @ bufH
  zerof<<<2048, 256, 0, stream>>>(bufF, (size_t)N * H);
  spmm_seg<128, SEG><<<sblocks, 256, 0, stream>>>(erow, ecol, ev, bufH, bufF, E);
  // 3) bufH = f16( relu(bufF) @ Wh0^T )
  gemm_wmma<128, 8, true><<<gblocks, 128, 0, stream>>>(bufF, Wh0h, bufH, mtiles);
  // 4) bufF = A_hat @ bufH
  zerof<<<2048, 256, 0, stream>>>(bufF, (size_t)N * H);
  spmm_seg<128, SEG><<<sblocks, 256, 0, stream>>>(erow, ecol, ev, bufH, bufF, E);
  // 5) bufH[:, :64] = f16( relu(bufF) @ W1^T )
  gemm_wmma<128, 4, true><<<gblocks, 128, 0, stream>>>(bufF, W1h, bufH, mtiles);
  // 6) d_out = A_hat @ bufH   (F = 64)
  zerof<<<2048, 256, 0, stream>>>(outp, (size_t)N * F);
  spmm_seg<64, SEG><<<sblocks, 256, 0, stream>>>(erow, ecol, ev, bufH, outp, E);
  // 7) row softmax
  softmax_rows64<<<(N + 7) / 8, 256, 0, stream>>>(outp, N);
}
